// GlobalMemoryNet_57509612093592
// MI455X (gfx1250) — compile-verified, matrix operand order checked by reading, not run on hardware
//
#include <hip/hip_runtime.h>
#include <math.h>

// ---- problem dims ----
#define BB   8
#define TT   16
#define NN   196
#define CC   768
#define HH   8
#define DD   96
#define HIDN 3072
#define ROWS (BB * NN)   // 1568
#define NP   256         // padded token/mem count for attention (16*16)

typedef __attribute__((ext_vector_type(16))) __bf16 v16bf;
typedef __attribute__((ext_vector_type(8)))  __bf16 v8bf;
typedef __attribute__((ext_vector_type(8)))  float  v8f;

// ---------------------------------------------------------------------------
// WMMA fragment loader: row-major bf16 source, 16 rows x 32 K per fragment.
// Lane l holds row (l&15); K elements {half*8..half*8+7} U {16+half*8..+7}
// where half = l>>4  (CDNA5 16-bit A-matrix 16x32 layout; B loaded from a
// transposed [N,K] array with the identical pattern).
// ---------------------------------------------------------------------------
__device__ __forceinline__ v16bf load_frag(const __bf16* base, int row0, int k0, int ld) {
    int lane = threadIdx.x & 31;
    int r    = lane & 15;
    int half = lane >> 4;
    const __bf16* p = base + (long)(row0 + r) * ld + k0 + half * 8;
    v8bf lo = *(const v8bf*)(p);
    v8bf hi = *(const v8bf*)(p + 16);   // +16 elements = K offset +16
    v16bf o;
#pragma unroll
    for (int i = 0; i < 8; ++i) { o[i] = lo[i]; o[i + 8] = hi[i]; }
    return o;
}

// ---------------------------------------------------------------------------
// Generic batched GEMM: D[m,n] = act( sum_k A[m,k]*Bt[n,k] + bias[n] )
// A: [M,K] bf16 row-major, Bt: [N,K] bf16 row-major (B transposed).
// Block = 8 waves (4x2); wave tile = 32 x (NT*16); block tile = 128 x (NT*32).
// act: 0=none, 1=exact GELU, 2=sigmoid(scale*x).  outF/outB nullable.
// batch z decomposed as zo=z/zInner, zi=z%zInner with separate strides.
// ---------------------------------------------------------------------------
template <int NT>
__global__ __launch_bounds__(256) void wmma_gemm(
    const __bf16* __restrict__ A, const __bf16* __restrict__ Bt,
    const float* __restrict__ bias,
    float* __restrict__ outF, __bf16* __restrict__ outB,
    int M, int N, int K, int ldOut, int Mstore,
    int zInner, long aSo, long aSi, long bSo, long bSi, long oSo, long oSi,
    int act, float scale)
{
    int wave = threadIdx.x >> 5;
    int lane = threadIdx.x & 31;
    int m0 = blockIdx.x * 128 + (wave >> 1) * 32;
    int n0 = blockIdx.y * (NT * 32) + (wave & 1) * (NT * 16);
    if (m0 >= M || n0 >= N) return;   // wave-uniform; EXEC stays all-1s

    int z  = blockIdx.z;
    int zo = z / zInner, zi = z % zInner;
    const __bf16* Ab = A  + zo * aSo + zi * aSi;
    const __bf16* Bb = Bt + zo * bSo + zi * bSi;
    long oBase = zo * oSo + zi * oSi;

    v8f acc[2][NT];
#pragma unroll
    for (int i = 0; i < 2; ++i)
#pragma unroll
        for (int j = 0; j < NT; ++j) acc[i][j] = (v8f){};

    for (int k0 = 0; k0 < K; k0 += 32) {
        v16bf a0 = load_frag(Ab, m0,      k0, K);
        v16bf a1 = load_frag(Ab, m0 + 16, k0, K);
        v16bf bf[NT];
#pragma unroll
        for (int t = 0; t < NT; ++t) bf[t] = load_frag(Bb, n0 + t * 16, k0, K);
#pragma unroll
        for (int t = 0; t < NT; ++t) {
            acc[0][t] = __builtin_amdgcn_wmma_f32_16x16x32_bf16(false, a0, false, bf[t], (short)0, acc[0][t], false, false);
            acc[1][t] = __builtin_amdgcn_wmma_f32_16x16x32_bf16(false, a1, false, bf[t], (short)0, acc[1][t], false, false);
        }
    }

    // C/D layout: VGPR r -> M = r + 8*half ; lane&15 -> N
    int col  = lane & 15;
    int half = lane >> 4;
#pragma unroll
    for (int tm = 0; tm < 2; ++tm) {
#pragma unroll
        for (int tn = 0; tn < NT; ++tn) {
            v8f av = acc[tm][tn];
            int n = n0 + tn * 16 + col;
            float bv = bias ? bias[n] : 0.0f;
#pragma unroll
            for (int r = 0; r < 8; ++r) {
                int m = m0 + tm * 16 + half * 8 + r;
                if (m < Mstore) {
                    float v = av[r] + bv;
                    if (act == 1)      v = 0.5f * v * (1.0f + erff(v * 0.70710678118654752f));
                    else if (act == 2) v = 1.0f / (1.0f + __expf(-v * scale));
                    long o = oBase + (long)m * ldOut + n;
                    if (outF) outF[o] = v;
                    if (outB) outB[o] = (__bf16)v;
                }
            }
        }
    }
}

// ---------------------------------------------------------------------------
// weight convert + transpose:  Wt[n][k] = (bf16) W[k][n]
// ---------------------------------------------------------------------------
__global__ void transpose_conv(const float* __restrict__ W, __bf16* __restrict__ Wt,
                               int K, int N)
{
    long idx = (long)blockIdx.x * 256 + threadIdx.x;
    long tot = (long)K * N;
    if (idx >= tot) return;
    long k = idx % K;
    long n = idx / K;
    Wt[idx] = (__bf16)W[k * (long)N + n];
}

__global__ void f32_to_bf16(const float* __restrict__ src, __bf16* __restrict__ dst, long n)
{
    long i = (long)blockIdx.x * 256 + threadIdx.x;
    if (i < n) dst[i] = (__bf16)src[i];
}

// ---------------------------------------------------------------------------
// LN of one frame of cur_fea; optionally writes f32 (init mem) and/or the
// bf16 first half of the concat buffer; also copies ap into the second half.
// blockIdx.x = row in [0,1568): b = row/196, n = row%196
// ---------------------------------------------------------------------------
__global__ __launch_bounds__(256) void ln_frame(
    const float* __restrict__ cur, int t,
    const float* __restrict__ w, const float* __restrict__ b,
    float* __restrict__ outF, __bf16* __restrict__ catB,
    const float* __restrict__ ap)
{
    int row = blockIdx.x;
    int bb  = row / NN, n = row % NN;
    const float* src = cur + (((long)bb * TT + t) * NN + n) * CC;
    int tid = threadIdx.x;

    float v0 = src[tid], v1 = src[tid + 256], v2 = src[tid + 512];
    __shared__ float ssum[256], ssq[256];
    ssum[tid] = v0 + v1 + v2;
    ssq[tid]  = v0 * v0 + v1 * v1 + v2 * v2;
    __syncthreads();
    for (int o = 128; o > 0; o >>= 1) {
        if (tid < o) { ssum[tid] += ssum[tid + o]; ssq[tid] += ssq[tid + o]; }
        __syncthreads();
    }
    float mean = ssum[0] * (1.0f / CC);
    float var  = ssq[0] * (1.0f / CC) - mean * mean;
    float rstd = rsqrtf(var + 1e-5f);

#pragma unroll
    for (int j = 0; j < 3; ++j) {
        int c = tid + j * 256;
        float v = (j == 0 ? v0 : (j == 1 ? v1 : v2));
        float o = (v - mean) * rstd * w[c] + b[c];
        if (outF) outF[(long)row * CC + c] = o;
        if (catB) {
            catB[(long)row * (2 * CC) + c] = (__bf16)o;
            catB[(long)row * (2 * CC) + CC + c] = (__bf16)ap[(long)bb * CC + c];
        }
    }
}

// ap[b][c] = mean_m mem[b][m][c]
__global__ void ap_kernel(const float* __restrict__ mem, float* __restrict__ ap)
{
    int idx = blockIdx.x * 256 + threadIdx.x;
    if (idx >= BB * CC) return;
    int bb = idx / CC, c = idx % CC;
    const float* p = mem + (long)bb * NN * CC + c;
    float s = 0.0f;
    for (int m = 0; m < NN; ++m) s += p[(long)m * CC];
    ap[idx] = s * (1.0f / NN);
}

// head-pack with zero pad: dst[bh][n][d] = src[(b*196+n)*768 + h*96+d], n<196
__global__ void repack_nd(const __bf16* __restrict__ src, __bf16* __restrict__ dst)
{
    long idx = (long)blockIdx.x * 256 + threadIdx.x;
    long tot = (long)BB * HH * NP * DD;
    if (idx >= tot) return;
    int d  = (int)(idx % DD);
    long r = idx / DD;
    int n  = (int)(r % NP);
    int bh = (int)(r / NP);
    int bb = bh >> 3, h = bh & 7;
    dst[idx] = (n < NN) ? src[((long)(bb * NN + n)) * CC + h * DD + d] : (__bf16)0.0f;
}

// transposed head-pack: dst[bh][d][n] = src[(b*196+n)*768 + h*96+d], n<196
__global__ void repack_dn(const __bf16* __restrict__ src, __bf16* __restrict__ dst)
{
    long idx = (long)blockIdx.x * 256 + threadIdx.x;
    long tot = (long)BB * HH * DD * NP;
    if (idx >= tot) return;
    int n  = (int)(idx % NP);
    long r = idx / NP;
    int d  = (int)(r % DD);
    int bh = (int)(r / DD);
    int bb = bh >> 3, h = bh & 7;
    dst[idx] = (n < NN) ? src[((long)(bb * NN + n)) * CC + h * DD + d] : (__bf16)0.0f;
}

// gate[bh][m] = 1 - mean_{n<196} ot[bh][m][n]
__global__ void gate_kernel(const __bf16* __restrict__ ot, float* __restrict__ gate)
{
    int idx = blockIdx.x * 256 + threadIdx.x;
    if (idx >= BB * HH * NN) return;
    int m  = idx % NN;
    int bh = idx / NN;
    const __bf16* p = ot + ((long)bh * NP + m) * NP;
    float s = 0.0f;
    for (int n = 0; n < NN; ++n) s += (float)p[n];
    gate[idx] = 1.0f - s * (1.0f / NN);
}

// new_mem = LN( p + gate[b, c/96, m] * memh )  (writes f32 dst)
__global__ __launch_bounds__(256) void final_ln(
    const float* __restrict__ pbuf, const float* __restrict__ gate,
    const float* __restrict__ memh,
    const float* __restrict__ w, const float* __restrict__ b,
    float* __restrict__ dst)
{
    int row = blockIdx.x;            // [0,1568)
    int bb  = row / NN, m = row % NN;
    int tid = threadIdx.x;

    float vals[3];
#pragma unroll
    for (int j = 0; j < 3; ++j) {
        int c = tid + j * 256;
        float g = gate[((long)bb * HH + (c / DD)) * NN + m];
        vals[j] = pbuf[(long)row * CC + c] + g * memh[(long)row * CC + c];
    }
    __shared__ float ssum[256], ssq[256];
    ssum[tid] = vals[0] + vals[1] + vals[2];
    ssq[tid]  = vals[0] * vals[0] + vals[1] * vals[1] + vals[2] * vals[2];
    __syncthreads();
    for (int o = 128; o > 0; o >>= 1) {
        if (tid < o) { ssum[tid] += ssum[tid + o]; ssq[tid] += ssq[tid + o]; }
        __syncthreads();
    }
    float mean = ssum[0] * (1.0f / CC);
    float var  = ssq[0] * (1.0f / CC) - mean * mean;
    float rstd = rsqrtf(var + 1e-5f);
#pragma unroll
    for (int j = 0; j < 3; ++j) {
        int c = tid + j * 256;
        dst[(long)row * CC + c] = (vals[j] - mean) * rstd * w[c] + b[c];
    }
}

// ---------------------------------------------------------------------------
// host side
// ---------------------------------------------------------------------------
static inline void launch_gemm(hipStream_t s, int nt,
    const __bf16* A, const __bf16* Bt, const float* bias,
    float* outF, __bf16* outB,
    int M, int N, int K, int ldOut, int Mstore,
    int nz, int zInner, long aSo, long aSi, long bSo, long bSi, long oSo, long oSi,
    int act, float scale)
{
    if (nt == 4) {
        dim3 g((M + 127) / 128, (N + 127) / 128, nz);
        wmma_gemm<4><<<g, 256, 0, s>>>(A, Bt, bias, outF, outB, M, N, K, ldOut, Mstore,
                                       zInner, aSo, aSi, bSo, bSi, oSo, oSi, act, scale);
    } else {
        dim3 g((M + 127) / 128, (N + 63) / 64, nz);
        wmma_gemm<2><<<g, 256, 0, s>>>(A, Bt, bias, outF, outB, M, N, K, ldOut, Mstore,
                                       zInner, aSo, aSi, bSo, bSi, oSo, oSi, act, scale);
    }
}

extern "C" void kernel_launch(void* const* d_in, const int* in_sizes, int n_in,
                              void* d_out, int out_size, void* d_ws, size_t ws_size,
                              hipStream_t stream)
{
    const float* cur  = (const float*)d_in[0];
    const float* n1w  = (const float*)d_in[1];
    const float* n1b  = (const float*)d_in[2];
    const float* n2w  = (const float*)d_in[3];
    const float* n2b  = (const float*)d_in[4];
    const float* c_w1 = (const float*)d_in[5];
    const float* c_b1 = (const float*)d_in[6];
    const float* c_w2 = (const float*)d_in[7];
    const float* c_b2 = (const float*)d_in[8];
    const float* m_w1 = (const float*)d_in[9];
    const float* m_b1 = (const float*)d_in[10];
    const float* m_w2 = (const float*)d_in[11];
    const float* m_b2 = (const float*)d_in[12];
    const float* p_w1 = (const float*)d_in[13];
    const float* p_b1 = (const float*)d_in[14];
    const float* p_w2 = (const float*)d_in[15];
    const float* p_b2 = (const float*)d_in[16];
    const float* q_w  = (const float*)d_in[17];
    const float* q_b  = (const float*)d_in[18];
    const float* k_w  = (const float*)d_in[19];
    const float* k_b  = (const float*)d_in[20];

    char* base = (char*)d_ws;
    size_t off = 0;
    auto carve = [&](size_t bytes) -> char* {
        char* p = base + off;
        off = (off + bytes + 255) & ~(size_t)255;
        return p;
    };

    // bf16 transposed weights [N][K]
    __bf16* cw1t = (__bf16*)carve((size_t)HIDN * 2 * CC * 2);
    __bf16* cw2t = (__bf16*)carve((size_t)CC * HIDN * 2);
    __bf16* mw1t = (__bf16*)carve((size_t)HIDN * CC * 2);
    __bf16* mw2t = (__bf16*)carve((size_t)CC * HIDN * 2);
    __bf16* pw1t = (__bf16*)carve((size_t)HIDN * CC * 2);
    __bf16* pw2t = (__bf16*)carve((size_t)CC * HIDN * 2);
    __bf16* qwt  = (__bf16*)carve((size_t)CC * CC * 2);
    __bf16* kwt  = (__bf16*)carve((size_t)CC * CC * 2);

    float*  memf  = (float*)carve((size_t)ROWS * CC * 4);
    __bf16* membf = (__bf16*)carve((size_t)ROWS * CC * 2);
    float*  apb   = (float*)carve((size_t)BB * CC * 4);
    __bf16* catb  = (__bf16*)carve((size_t)ROWS * 2 * CC * 2);
    __bf16* hid   = (__bf16*)carve((size_t)ROWS * HIDN * 2);
    __bf16* cgbf  = (__bf16*)carve((size_t)ROWS * CC * 2);
    float*  memh  = (float*)carve((size_t)ROWS * CC * 4);
    __bf16* qbf   = (__bf16*)carve((size_t)ROWS * CC * 2);
    __bf16* kbf   = (__bf16*)carve((size_t)ROWS * CC * 2);
    __bf16* qp    = (__bf16*)carve((size_t)BB * HH * NP * DD * 2);
    __bf16* kp    = (__bf16*)carve((size_t)BB * HH * NP * DD * 2);
    __bf16* cgT   = (__bf16*)carve((size_t)BB * HH * DD * NP * 2);
    __bf16* ot    = (__bf16*)carve((size_t)BB * HH * NP * NP * 2);
    float*  gateb = (float*)carve((size_t)BB * HH * NN * 4);
    float*  cf    = (float*)carve((size_t)ROWS * CC * 4);
    __bf16* cbf   = (__bf16*)carve((size_t)ROWS * CC * 2);
    float*  pf    = (float*)carve((size_t)ROWS * CC * 4);

    auto tc = [&](const float* W, __bf16* Wt, int K, int N) {
        long tot = (long)K * N;
        transpose_conv<<<(unsigned)((tot + 255) / 256), 256, 0, stream>>>(W, Wt, K, N);
    };
    tc(c_w1, cw1t, 2 * CC, HIDN);
    tc(c_w2, cw2t, HIDN, CC);
    tc(m_w1, mw1t, CC, HIDN);
    tc(m_w2, mw2t, HIDN, CC);
    tc(p_w1, pw1t, CC, HIDN);
    tc(p_w2, pw2t, HIDN, CC);
    tc(q_w,  qwt,  CC, CC);
    tc(k_w,  kwt,  CC, CC);

    // init memory = LN(frame 0)
    ln_frame<<<ROWS, 256, 0, stream>>>(cur, 0, n1w, n1b, memf, nullptr, nullptr);

    const float scale = 0.10206207261596575f; // 96^-0.5
    const long ew = (long)ROWS * CC;
    const long packND = (long)BB * HH * NP * DD;

    for (int t = 1; t < TT; ++t) {
        ap_kernel<<<(BB * CC + 255) / 256, 256, 0, stream>>>(memf, apb);
        ln_frame<<<ROWS, 256, 0, stream>>>(cur, t, n1w, n1b, nullptr, catb, apb);
        f32_to_bf16<<<(unsigned)((ew + 255) / 256), 256, 0, stream>>>(memf, membf, ew);

        // c_func MLP: cat[1568,1536] -> GELU -> cg[1568,768]
        launch_gemm(stream, 4, catb, cw1t, c_b1, nullptr, hid, ROWS, HIDN, 2 * CC, HIDN, ROWS,
                    1, 1, 0, 0, 0, 0, 0, 0, /*act=*/1, 0.f);
        launch_gemm(stream, 4, hid, cw2t, c_b2, nullptr, cgbf, ROWS, CC, HIDN, CC, ROWS,
                    1, 1, 0, 0, 0, 0, 0, 0, 0, 0.f);
        // m_func MLP: mem -> memh (f32)
        launch_gemm(stream, 4, membf, mw1t, m_b1, nullptr, hid, ROWS, HIDN, CC, HIDN, ROWS,
                    1, 1, 0, 0, 0, 0, 0, 0, 1, 0.f);
        launch_gemm(stream, 4, hid, mw2t, m_b2, memh, nullptr, ROWS, CC, HIDN, CC, ROWS,
                    1, 1, 0, 0, 0, 0, 0, 0, 0, 0.f);
        // q/k projections
        launch_gemm(stream, 4, cgbf,  qwt, q_b, nullptr, qbf, ROWS, CC, CC, CC, ROWS,
                    1, 1, 0, 0, 0, 0, 0, 0, 0, 0.f);
        launch_gemm(stream, 4, membf, kwt, k_b, nullptr, kbf, ROWS, CC, CC, CC, ROWS,
                    1, 1, 0, 0, 0, 0, 0, 0, 0, 0.f);

        repack_nd<<<(unsigned)((packND + 255) / 256), 256, 0, stream>>>(qbf, qp);
        repack_nd<<<(unsigned)((packND + 255) / 256), 256, 0, stream>>>(kbf, kp);
        repack_dn<<<(unsigned)((packND + 255) / 256), 256, 0, stream>>>(cgbf, cgT);

        // ot[bh][m][n] = sigmoid(scale * sum_d k[m,d]*q[n,d]) == o[b,h,n,m]
        launch_gemm(stream, 4, kp, qp, nullptr, nullptr, ot, NP, NP, DD, NP, NP,
                    BB * HH, 1, (long)NP * DD, 0, (long)NP * DD, 0, (long)NP * NP, 0,
                    /*act=*/2, scale);

        gate_kernel<<<(BB * HH * NN + 255) / 256, 256, 0, stream>>>(ot, gateb);

        // c[b,m,h*96+d] = sum_n ot[bh][m][n] * cgT[bh][d][n]   (store m<196)
        launch_gemm(stream, 2, ot, cgT, nullptr, cf, cbf, NP, DD, NP, CC, NN,
                    BB * HH, HH,
                    (long)HH * NP * NP, (long)NP * NP,
                    (long)HH * DD * NP, (long)DD * NP,
                    (long)NN * CC, (long)DD,
                    0, 0.f);

        // post MLP
        launch_gemm(stream, 4, cbf, pw1t, p_b1, nullptr, hid, ROWS, HIDN, CC, HIDN, ROWS,
                    1, 1, 0, 0, 0, 0, 0, 0, 1, 0.f);
        launch_gemm(stream, 4, hid, pw2t, p_b2, pf, nullptr, ROWS, CC, HIDN, CC, ROWS,
                    1, 1, 0, 0, 0, 0, 0, 0, 0, 0.f);

        float* dst = (t == TT - 1) ? (float*)d_out : memf;
        final_ln<<<ROWS, 256, 0, stream>>>(pf, gateb, memh, n2w, n2b, dst);
    }
}
